// summary_content_selector_56100862820941
// MI455X (gfx1250) — matrix operand order repaired and checked by reading, not compiled, not run on hardware
//
#include <hip/hip_runtime.h>
#include <hip/hip_bf16.h>

// ---------- problem constants (fixed by the reference) ----------
constexpr int Bc = 16;    // batch
constexpr int Tc = 1024;  // target len
constexpr int Sc = 2048;  // source len
constexpr int Dc = 1024;  // dim
constexpr int BT = Bc * Tc;

// ---------- vector types ----------
typedef __attribute__((ext_vector_type(8)))  float           v8f;
typedef __attribute__((ext_vector_type(16))) __bf16          v16bf;
typedef __attribute__((ext_vector_type(8)))  unsigned short  ush8;
typedef __attribute__((ext_vector_type(16))) unsigned short  ush16;
typedef __attribute__((ext_vector_type(4)))  unsigned short  ush4;
typedef __attribute__((ext_vector_type(4)))  float           fl4;
// exact pointee type expected by the async-LDS builtin (per compiler diagnostic)
typedef int i4v __attribute__((vector_size(4 * sizeof(int))));

// ---------- CDNA5 async global->LDS path (guarded; sync fallback) ----------
#if __has_builtin(__builtin_amdgcn_global_load_async_to_lds_b128)
#define ASYNC_LDS 1
#else
#define ASYNC_LDS 0
#endif

__device__ __forceinline__ void cp16_to_lds(const unsigned short* g, unsigned short* l) {
#if ASYNC_LDS
    // per-lane 16B async copy, tracked by ASYNCcnt (GLOBAL_LOAD_ASYNC_TO_LDS_B128)
    __builtin_amdgcn_global_load_async_to_lds_b128(
        (__attribute__((address_space(1))) i4v*)(unsigned long long)g,
        (__attribute__((address_space(3))) i4v*)(unsigned long long)l,
        0, 0);
#else
    *(ush8*)l = *(const ush8*)g;
#endif
}

__device__ __forceinline__ void wait_async_lds() {
#if ASYNC_LDS
#if __has_builtin(__builtin_amdgcn_s_wait_asynccnt)
    __builtin_amdgcn_s_wait_asynccnt(0);
#else
    asm volatile("s_wait_asynccnt 0x0" ::: "memory");
#endif
#endif
}

__device__ __forceinline__ unsigned short f2bf(float f) {
    unsigned u = __builtin_bit_cast(unsigned, f);
    u = (u + 0x7FFFu + ((u >> 16) & 1u)) >> 16;   // round-to-nearest-even
    return (unsigned short)u;
}

// ---------- fp32 -> bf16 convert (vectorized x4, bandwidth bound) ----------
__global__ __launch_bounds__(256) void k_cvt_bf16(const float* __restrict__ x,
                                                  unsigned short* __restrict__ y,
                                                  int n4) {
    int i = blockIdx.x * 256 + threadIdx.x;
    if (i < n4) {
        fl4 f = ((const fl4*)x)[i];
        ush4 o;
        o.x = f2bf(f.x); o.y = f2bf(f.y); o.z = f2bf(f.z); o.w = f2bf(f.w);
        ((ush4*)y)[i] = o;
    }
}

// copy source (fp32 [BT,D]) into second half of concat buffer (bf16 [BT,2D])
__global__ __launch_bounds__(256) void k_src_to_concat(const float* __restrict__ src,
                                                       unsigned short* __restrict__ concat) {
    int i4 = blockIdx.x * 256 + threadIdx.x;          // 4-elem chunk id
    int i = i4 * 4;
    int m = i >> 10;                                   // / Dc
    int d = i & (Dc - 1);
    fl4 f = *(const fl4*)(src + i);
    ush4 o;
    o.x = f2bf(f.x); o.y = f2bf(f.y); o.z = f2bf(f.z); o.w = f2bf(f.w);
    *(ush4*)(concat + (size_t)m * (2 * Dc) + Dc + d) = o;
}

// ---------- tiled bf16 WMMA GEMM, double-buffered LDS ----------
// Block = 256 threads = 8 wave32 waves laid out 4(M) x 2(N).
// Block tile 128(M) x 64(N), K step 32. Each wave owns 32x32 = 2x2 WMMA tiles.
// EPI: 0 = store bf16 h_t          (C = ht[m*Dc + n])
//      1 = mask + fp32 scores      (C = align[t*B*S + bz*S + n], t=m)
//      2 = store bf16 c -> concat  (C = concat[(bz*T+m)*2D + n])
//      3 = tanh fp32 attn_h        (C = attn[(m%T)*B*D + (m/T)*D + n])
// BNK: true  -> B operand given as [N,K] rows (contiguous K per n)
//      false -> B operand given as [K,N] rows (transpose while staging to LDS)
template <int EPI, bool BNK>
__global__ __launch_bounds__(256)
void k_gemm_bf16(const unsigned short* __restrict__ A, int lda, long long aBatch,
                 const unsigned short* __restrict__ Bm, int ldb, long long bBatch,
                 void* __restrict__ Cp, const int* __restrict__ ml,
                 int M, int N, int K) {
    __shared__ unsigned short ldsA[2][128][40];   // 32 K + 8 pad, double buffered
    __shared__ unsigned short ldsB[2][64][40];

    const int tid = threadIdx.x;
    const int bz  = blockIdx.z;
    A  += (size_t)bz * aBatch;
    Bm += (size_t)bz * bBatch;

    const int m0 = blockIdx.y * 128;
    const int n0 = blockIdx.x * 64;

    const int lane = tid & 31;
    const int w    = tid >> 5;
    const int wm   = (w & 3) * 32;     // wave M offset in block tile
    const int wn   = (w >> 2) * 32;    // wave N offset in block tile
    const int h    = lane >> 4;        // K-half selector per ISA layout
    const int lm   = lane & 15;

    v8f acc[2][2] = {};

    // ---- tile staging helpers ----
    auto stageA = [&](int k0, int buf) {
#pragma unroll
        for (int it = 0; it < 2; ++it) {
            int c  = tid + it * 256;
            int r  = c >> 2;
            int kc = (c & 3) << 3;
            cp16_to_lds(A + (size_t)(m0 + r) * lda + k0 + kc, &ldsA[buf][r][kc]);
        }
    };
    auto stageB = [&](int k0, int buf) {
        if constexpr (BNK) {
            int r  = tid >> 2;
            int kc = (tid & 3) << 3;
            cp16_to_lds(Bm + (size_t)(n0 + r) * ldb + k0 + kc, &ldsB[buf][r][kc]);
        } else {
            // B given [K,N]: read coalesced along n, transpose into LDS (sync)
#pragma unroll
            for (int kk = 0; kk < 4; ++kk) {
                int k  = kk * 8 + (tid >> 5);
                int np = (tid & 31) << 1;
                unsigned v = *(const unsigned*)(Bm + (size_t)(k0 + k) * ldb + n0 + np);
                ldsB[buf][np][k]     = (unsigned short)(v & 0xFFFFu);
                ldsB[buf][np + 1][k] = (unsigned short)(v >> 16);
            }
        }
    };

    const int nk = K >> 5;             // K / 32 steps
    stageA(0, 0);
    stageB(0, 0);
    wait_async_lds();
    __syncthreads();

    for (int ki = 0; ki < nk; ++ki) {
        const int cur = ki & 1;
        const int nxt = cur ^ 1;

        // prefetch-and-stage the next tile while computing on the current one
        if (ki + 1 < nk) {
            stageA((ki + 1) << 5, nxt);
            stageB((ki + 1) << 5, nxt);
        }
        if (ki + 2 < nk)
            __builtin_prefetch(A + (size_t)(m0 + (tid >> 1)) * lda + ((ki + 2) << 5), 0, 3);

        // ---- build fragments per ISA 16-bit A(16x32)/B(32x16) lane layouts ----
        v16bf aF[2], bF[2];
#pragma unroll
        for (int mi = 0; mi < 2; ++mi) {
            int row = wm + mi * 16 + lm;
            ush8 lo = *(const ush8*)&ldsA[cur][row][h * 8];       // K {0..7} / {8..15}
            ush8 hi = *(const ush8*)&ldsA[cur][row][16 + h * 8];  // K {16..23} / {24..31}
            ush16 u = __builtin_shufflevector(lo, hi, 0,1,2,3,4,5,6,7,
                                              8,9,10,11,12,13,14,15);
            aF[mi] = __builtin_bit_cast(v16bf, u);
        }
#pragma unroll
        for (int ni = 0; ni < 2; ++ni) {
            int row = wn + ni * 16 + lm;
            ush8 lo = *(const ush8*)&ldsB[cur][row][h * 8];
            ush8 hi = *(const ush8*)&ldsB[cur][row][16 + h * 8];
            ush16 u = __builtin_shufflevector(lo, hi, 0,1,2,3,4,5,6,7,
                                              8,9,10,11,12,13,14,15);
            bF[ni] = __builtin_bit_cast(v16bf, u);
        }
#pragma unroll
        for (int mi = 0; mi < 2; ++mi)
#pragma unroll
            for (int ni = 0; ni < 2; ++ni)
                acc[mi][ni] = __builtin_amdgcn_wmma_f32_16x16x32_bf16(
                    false, aF[mi], false, bF[ni], (short)0, acc[mi][ni],
                    false, false);

        // our async writes to 'nxt' must land before the cross-wave barrier
        wait_async_lds();
        __syncthreads();
    }

    // ---- epilogue ----
    int len = 0;
    if (EPI == 1) len = ml[bz];

#pragma unroll
    for (int mi = 0; mi < 2; ++mi)
#pragma unroll
        for (int ni = 0; ni < 2; ++ni) {
            int n_g = n0 + wn + ni * 16 + lm;
#pragma unroll
            for (int i = 0; i < 8; ++i) {
                int m_g = m0 + wm + mi * 16 + h * 8 + i;
                float v = acc[mi][ni][i];
                if (EPI == 0) {
                    ((unsigned short*)Cp)[(size_t)m_g * Dc + n_g] = f2bf(v);
                } else if (EPI == 1) {
                    ((float*)Cp)[(size_t)m_g * (Bc * Sc) + (size_t)bz * Sc + n_g] =
                        (n_g < len) ? v : -1e9f;
                } else if (EPI == 2) {
                    ((unsigned short*)Cp)[((size_t)bz * Tc + m_g) * (2 * Dc) + n_g] =
                        f2bf(v);
                } else {
                    int t = m_g & (Tc - 1);
                    int b = m_g >> 10;   // / Tc
                    ((float*)Cp)[(size_t)t * (Bc * Dc) + (size_t)b * Dc + n_g] =
                        tanhf(v);
                }
            }
        }
}

// ---------- row softmax over S, in-place fp32 + bf16 copy at [B,T,S] ----------
__global__ __launch_bounds__(256)
void k_softmax(float* __restrict__ alignF, unsigned short* __restrict__ pbf) {
    const int row = blockIdx.x;           // row = t*B + b  ([T,B,S] layout)
    const int t = row >> 4, b = row & (Bc - 1);
    float* p = alignF + (size_t)row * Sc;
    __shared__ float red[256];
    const int tid = threadIdx.x;

    float vals[8];
    float mx = -3.4e38f;
#pragma unroll
    for (int j = 0; j < 8; ++j) {
        vals[j] = p[tid + j * 256];
        mx = fmaxf(mx, vals[j]);
    }
    red[tid] = mx; __syncthreads();
    for (int s = 128; s > 0; s >>= 1) {
        if (tid < s) red[tid] = fmaxf(red[tid], red[tid + s]);
        __syncthreads();
    }
    mx = red[0]; __syncthreads();

    float sum = 0.f;
#pragma unroll
    for (int j = 0; j < 8; ++j) {
        vals[j] = __expf(vals[j] - mx);
        sum += vals[j];
    }
    red[tid] = sum; __syncthreads();
    for (int s = 128; s > 0; s >>= 1) {
        if (tid < s) red[tid] += red[tid + s];
        __syncthreads();
    }
    const float inv = 1.0f / red[0];

    unsigned short* q = pbf + ((size_t)b * Tc + t) * Sc;
#pragma unroll
    for (int j = 0; j < 8; ++j) {
        int s = tid + j * 256;
        float r = vals[j] * inv;
        p[s] = r;
        q[s] = f2bf(r);
    }
}

// ---------- host launcher ----------
extern "C" void kernel_launch(void* const* d_in, const int* in_sizes, int n_in,
                              void* d_out, int out_size, void* d_ws, size_t ws_size,
                              hipStream_t stream) {
    (void)in_sizes; (void)n_in; (void)out_size; (void)ws_size;
    const float* src  = (const float*)d_in[0];   // [B,T,D]
    const float* memb = (const float*)d_in[1];   // [B,S,D]
    const int*   mlen = (const int*)d_in[2];     // [B]
    const float* Wi   = (const float*)d_in[3];   // [D,D]
    const float* Wo   = (const float*)d_in[4];   // [D,2D]

    float* attn   = (float*)d_out;                                  // [T,B,D]
    float* alignF = (float*)d_out + (size_t)Tc * Bc * Dc;           // [T,B,S]

    // workspace carve-up (bf16 buffers)
    unsigned short* concatB = (unsigned short*)d_ws;                 // [BT, 2D]
    unsigned short* memB    = concatB + (size_t)BT * 2 * Dc;         // [B*S, D]
    unsigned short* htB     = memB    + (size_t)Bc * Sc * Dc;        // [BT, D]
    unsigned short* pB      = htB     + (size_t)BT * Dc;             // [B,T,S]
    unsigned short* WiB     = pB      + (size_t)Bc * Tc * Sc;        // [D, D]
    unsigned short* WoB     = WiB     + (size_t)Dc * Dc;             // [D, 2D]

    // 1) precision conversion (bandwidth bound)
    k_cvt_bf16<<<(Bc * Sc * Dc / 4 + 255) / 256, 256, 0, stream>>>(memb, memB, Bc * Sc * Dc / 4);
    k_cvt_bf16<<<(Dc * Dc / 4 + 255) / 256, 256, 0, stream>>>(Wi, WiB, Dc * Dc / 4);
    k_cvt_bf16<<<(Dc * 2 * Dc / 4 + 255) / 256, 256, 0, stream>>>(Wo, WoB, Dc * 2 * Dc / 4);
    k_src_to_concat<<<BT * Dc / 4 / 256, 256, 0, stream>>>(src, concatB);

    // 2) h_t = src @ W_in^T     (A = concat second half, lda = 2D)
    k_gemm_bf16<0, true><<<dim3(Dc / 64, BT / 128, 1), 256, 0, stream>>>(
        concatB + Dc, 2 * Dc, 0, WiB, Dc, 0, htB, nullptr, BT, Dc, Dc);

    // 3) scores = h_t @ mem^T, masked, written [T,B,S]
    k_gemm_bf16<1, true><<<dim3(Sc / 64, Tc / 128, Bc), 256, 0, stream>>>(
        htB, Dc, (long long)Tc * Dc, memB, Dc, (long long)Sc * Dc,
        alignF, mlen, Tc, Sc, Dc);

    // 4) softmax rows (fp32 in place -> output; bf16 copy [B,T,S] for next GEMM)
    k_softmax<<<Tc * Bc, 256, 0, stream>>>(alignF, pB);

    // 5) c = p @ mem   (B operand is [K=S, N=D]: transpose while staging)
    k_gemm_bf16<2, false><<<dim3(Dc / 64, Tc / 128, Bc), 256, 0, stream>>>(
        pB, Sc, (long long)Tc * Sc, memB, Dc, (long long)Sc * Dc,
        concatB, nullptr, Tc, Dc, Sc);

    // 6) attn_h = tanh(concat @ W_out^T), written [T,B,D]
    k_gemm_bf16<3, true><<<dim3(Dc / 64, BT / 128, 1), 256, 0, stream>>>(
        concatB, 2 * Dc, 0, WoB, 2 * Dc, 0, attn, nullptr, BT, Dc, 2 * Dc);
}
